// Attention_68066641707351
// MI455X (gfx1250) — compile-verified, matrix operand order checked by reading, not run on hardware
//
#include <hip/hip_runtime.h>

typedef __attribute__((ext_vector_type(16))) __bf16 v16bf;
typedef __attribute__((ext_vector_type(8)))  __bf16 v8bf;
typedef __attribute__((ext_vector_type(8)))  float  v8f;

#define DIMC 512
#define NTOK 4096
#define NB   4
#define SP   4100              // padded f32 row stride for score tile
#define SMEM2 (16 * SP * 4)    // 262400 bytes dynamic LDS

#define LOCAL_AS __attribute__((address_space(3)))

// ---- fragment helpers -------------------------------------------------------
// A(16x32 bf16) per-lane layout: lane<16 -> row=lane, K in {0..7}U{16..23};
// lane>=16 -> row=lane-16, K in {8..15}U{24..31}.  B(32x16) mirrors with N=lane&15.
__device__ __forceinline__ v16bf ldfrag(const __bf16* p) {
  v8bf lo = *(const v8bf*)p;        // K + {0..7}
  v8bf hi = *(const v8bf*)(p + 16); // K + {16..23}
  return __builtin_shufflevector(lo, hi, 0,1,2,3,4,5,6,7,8,9,10,11,12,13,14,15);
}

__device__ __forceinline__ void cvt8(v16bf& A, int base, float4 u, float4 v) {
  A[base+0] = (__bf16)u.x; A[base+1] = (__bf16)u.y;
  A[base+2] = (__bf16)u.z; A[base+3] = (__bf16)u.w;
  A[base+4] = (__bf16)v.x; A[base+5] = (__bf16)v.y;
  A[base+6] = (__bf16)v.z; A[base+7] = (__bf16)v.w;
}

// ---- kernel 0: cast weights to bf16 ----------------------------------------
__global__ void cast_weights(const float* __restrict__ w1,
                             const float* __restrict__ w2,
                             __bf16* __restrict__ w1b,
                             __bf16* __restrict__ w2b) {
  int i = blockIdx.x * blockDim.x + threadIdx.x;   // 0 .. 512*512-1
  w1b[i] = (__bf16)w1[i];
  w2b[i] = (__bf16)w2[i];
}

// ---- kernel 1: f -> fb (bf16), q = f*w1^T, vT = (f*w2^T)^T -----------------
__global__ __launch_bounds__(512, 1)
void project_kernel(const float* __restrict__ x,
                    const __bf16* __restrict__ w1b,
                    const __bf16* __restrict__ w2b,
                    __bf16* __restrict__ fb,     // [b][tok][c]
                    __bf16* __restrict__ q16,    // [b][tok][d]
                    __bf16* __restrict__ vT)     // [b][d][tok]
{
  __shared__ float  Xs[DIMC][20];                // raw f32 tile, [c][tok] (+pad)
  __shared__ __bf16 Fs[16][520];                 // bf16 tile, [tok][c] (+pad)

  const int b  = blockIdx.y;
  const int n0 = blockIdx.x * 16;
  const int c  = threadIdx.x;                    // 0..511

  // Async DMA: lane c pulls x[b][c][n0..n0+15] (64B) straight into LDS,
  // bypassing VGPRs (gfx1250 GLOBAL_LOAD_ASYNC_TO_LDS, tracked by ASYNCcnt).
  {
    const float* src = x + ((size_t)b * DIMC + c) * NTOK + n0;
    unsigned lds0 = (unsigned)(uintptr_t)(LOCAL_AS float*)&Xs[c][0];
    #pragma unroll
    for (int j = 0; j < 4; ++j) {
      unsigned dst = lds0 + j * 16;
      unsigned long long ga = (unsigned long long)(uintptr_t)(src + j * 4);
      asm volatile("global_load_async_to_lds_b128 %0, %1, off"
                   :: "v"(dst), "v"(ga) : "memory");
    }
    asm volatile("s_wait_asynccnt 0x0" ::: "memory");
  }

  // convert own column to bf16: fill Fs[tok][c] and emit fb (K matrix) rows
  #pragma unroll
  for (int r = 0; r < 16; ++r) {
    __bf16 v = (__bf16)Xs[c][r];
    Fs[r][c] = v;
    fb[((size_t)b * NTOK + n0 + r) * DIMC + c] = v;
  }
  __syncthreads();

  const int w   = threadIdx.x >> 5;
  const int l   = threadIdx.x & 31;
  const int col = l & 15;
  const int grp = (l >> 4) ? 8 : 0;

  #pragma unroll
  for (int rep = 0; rep < 2; ++rep) {
    const int dt = w + rep * 16;                 // d-tile (0..31)
    v8f accq = {}; v8f accv = {};
    #pragma unroll
    for (int kk = 0; kk < 16; ++kk) {
      v16bf A  = ldfrag(&Fs[col][kk * 32 + grp]);
      v16bf B1 = ldfrag(w1b + (size_t)(dt * 16 + col) * DIMC + kk * 32 + grp);
      v16bf B2 = ldfrag(w2b + (size_t)(dt * 16 + col) * DIMC + kk * 32 + grp);
      accq = __builtin_amdgcn_wmma_f32_16x16x32_bf16(false, A, false, B1, (short)0, accq, false, false);
      accv = __builtin_amdgcn_wmma_f32_16x16x32_bf16(false, A, false, B2, (short)0, accv, false, false);
    }
    #pragma unroll
    for (int i = 0; i < 8; ++i) {
      const int M = i + grp;                     // token row in tile
      q16[((size_t)b * NTOK + n0 + M) * DIMC + dt * 16 + col] = (__bf16)accq[i];
      vT [((size_t)b * DIMC + dt * 16 + col) * NTOK + n0 + M] = (__bf16)accv[i];
    }
  }
}

// ---- kernel 2: attention for one 16-query tile ------------------------------
__global__ __launch_bounds__(512, 1)
void attn_kernel(const __bf16* __restrict__ q16,
                 const __bf16* __restrict__ fb,
                 const __bf16* __restrict__ vT,
                 float* __restrict__ out)
{
  extern __shared__ float S[];                   // [16][SP] f32 scores
  __shared__ float invsum[16];

  const int b  = blockIdx.y;
  const int m0 = blockIdx.x * 16;
  const int w   = threadIdx.x >> 5;
  const int l   = threadIdx.x & 31;
  const int col = l & 15;
  const int grp = (l >> 4) ? 8 : 0;
  const float scale = 0.044194173824159216f;     // 512^-0.5

  // preload all 16 Q A-fragments (row = col, K = kk*32 + grp + ...)
  const __bf16* qrow = q16 + ((size_t)b * NTOK + m0 + col) * DIMC;
  v16bf qa[16];
  #pragma unroll
  for (int kk = 0; kk < 16; ++kk)
    qa[kk] = ldfrag(qrow + kk * 32 + grp);

  // ---- phase 1: S = scale * Q K^T  (each wave: 16 column tiles) ----
  for (int j = 0; j < 16; ++j) {
    const int ct = j * 16 + w;                   // 0..255
    const int nb = ct * 16;
    v8f acc = {};
    const __bf16* kb = fb + ((size_t)b * NTOK + nb + col) * DIMC;
    if (j < 15)
      __builtin_prefetch(kb + 256 * DIMC, 0, 0); // next column tile (L2 stream)
    #pragma unroll
    for (int kk = 0; kk < 16; ++kk) {
      v16bf B = ldfrag(kb + kk * 32 + grp);
      acc = __builtin_amdgcn_wmma_f32_16x16x32_bf16(false, qa[kk], false, B, (short)0, acc, false, false);
    }
    #pragma unroll
    for (int i = 0; i < 8; ++i)
      S[(size_t)(i + grp) * SP + nb + col] = acc[i] * scale;
  }
  __syncthreads();

  // ---- phase 2: row softmax, wave w owns row w ----
  {
    float* Srow = &S[(size_t)w * SP];
    float m = -3.0e38f;
    for (int n = l; n < NTOK; n += 32) m = fmaxf(m, Srow[n]);
    #pragma unroll
    for (int off = 16; off > 0; off >>= 1) m = fmaxf(m, __shfl_xor(m, off, 32));
    float s = 0.f;
    for (int n = l; n < NTOK; n += 32) {
      float e = __expf(Srow[n] - m);
      Srow[n] = e;
      s += e;
    }
    #pragma unroll
    for (int off = 16; off > 0; off >>= 1) s += __shfl_xor(s, off, 32);
    if (l == 0) invsum[w] = 1.0f / s;
  }
  __syncthreads();

  // ---- phase 3: O = P V  (each wave: 2 d-tiles, K = 4096) ----
  #pragma unroll
  for (int rep = 0; rep < 2; ++rep) {
    const int dt = w + rep * 16;
    v8f acc = {};
    const __bf16* vrow = vT + ((size_t)b * DIMC + dt * 16 + col) * NTOK;
    #pragma unroll 4
    for (int ks = 0; ks < 128; ++ks) {
      const float* sp = &S[(size_t)col * SP + ks * 32 + grp];
      float4 a0 = *(const float4*)(sp + 0);
      float4 a1 = *(const float4*)(sp + 4);
      float4 a2 = *(const float4*)(sp + 16);
      float4 a3 = *(const float4*)(sp + 20);
      v16bf A;
      cvt8(A, 0, a0, a1);
      cvt8(A, 8, a2, a3);
      v16bf B = ldfrag(vrow + ks * 32 + grp);
      acc = __builtin_amdgcn_wmma_f32_16x16x32_bf16(false, A, false, B, (short)0, acc, false, false);
    }
    #pragma unroll
    for (int i = 0; i < 8; ++i) {
      const int M = i + grp;
      out[((size_t)b * DIMC + dt * 16 + col) * NTOK + m0 + M] = acc[i] * invsum[M];
    }
  }
}

// ---- host launcher ----------------------------------------------------------
extern "C" void kernel_launch(void* const* d_in, const int* in_sizes, int n_in,
                              void* d_out, int out_size, void* d_ws, size_t ws_size,
                              hipStream_t stream) {
  const float* x  = (const float*)d_in[0];
  const float* w1 = (const float*)d_in[1];
  const float* w2 = (const float*)d_in[2];
  float* out = (float*)d_out;

  char* ws = (char*)d_ws;
  const size_t fbSz = (size_t)NB * NTOK * DIMC * 2;   // 16 MB
  __bf16* fb  = (__bf16*)(ws);
  __bf16* q16 = (__bf16*)(ws + fbSz);
  __bf16* vT  = (__bf16*)(ws + 2 * fbSz);
  __bf16* w1b = (__bf16*)(ws + 3 * fbSz);
  __bf16* w2b = (__bf16*)(ws + 3 * fbSz + (size_t)DIMC * DIMC * 2);

  (void)in_sizes; (void)n_in; (void)out_size; (void)ws_size;

  // idempotent, host-side, capture-safe; called unconditionally (no static guards)
  hipFuncSetAttribute((const void*)attn_kernel,
                      hipFuncAttributeMaxDynamicSharedMemorySize, SMEM2);

  cast_weights<<<dim3((DIMC * DIMC) / 256), dim3(256), 0, stream>>>(w1, w2, w1b, w2b);

  dim3 grid(NTOK / 16, NB);
  project_kernel<<<grid, dim3(512), 0, stream>>>(x, w1b, w2b, fb, q16, vT);
  attn_kernel<<<grid, dim3(512), SMEM2, stream>>>(q16, fb, vT, out);
}
